// ResidualGroup_6270652252947
// MI455X (gfx1250) — compile-verified, matrix operand order checked by reading, not run on hardware
//
#include <hip/hip_runtime.h>
#include <math.h>

// ---------------------------------------------------------------------------
// CDNA5 WMMA (wave32): v_wmma_f32_16x16x32_bf16
// ---------------------------------------------------------------------------
typedef __attribute__((ext_vector_type(16))) __bf16 v16bf;
typedef __attribute__((ext_vector_type(8)))  __bf16 v8bf;
typedef __attribute__((ext_vector_type(8)))  float  v8f;

__device__ __forceinline__ __bf16 f2bf(float f) {
  union { float f; unsigned int u; } x; x.f = f;
  unsigned int r = x.u + 0x7fffu + ((x.u >> 16) & 1u);   // round-to-nearest-even
  union { unsigned short s; __bf16 b; } y;
  y.s = (unsigned short)(r >> 16);
  return y.b;
}
__device__ __forceinline__ float bf2f(__bf16 b) {
  union { unsigned short s; __bf16 b; } x; x.b = b;
  union { unsigned int u; float f; } y; y.u = ((unsigned int)x.s) << 16;
  return y.f;
}
__device__ __forceinline__ __bf16 bfzero() {
  union { unsigned short s; __bf16 b; } x; x.s = 0; return x.b;
}
__device__ __forceinline__ v16bf cat8(v8bf a, v8bf b) {
  return __builtin_shufflevector(a, b, 0,1,2,3,4,5,6,7,8,9,10,11,12,13,14,15);
}

// ---------------------------------------------------------------------------
// fp32 -> bf16 elementwise (weights / input transpose)
// ---------------------------------------------------------------------------
__global__ void cvt_bf16_k(const float* __restrict__ In, __bf16* __restrict__ Out, int n)
{
  int i = blockIdx.x * blockDim.x + threadIdx.x;
  if (i < n) Out[i] = f2bf(In[i]);
}

// NCHW fp32 -> NHWC bf16
__global__ void nchw2nhwc_k(const float* __restrict__ In, __bf16* __restrict__ Out,
                            int Nb, int C, int H, int W)
{
  int idx = blockIdx.x * blockDim.x + threadIdx.x;
  int total = Nb * C * H * W;
  if (idx >= total) return;
  int c = idx % C;
  int t = idx / C;
  int w = t % W; t /= W;
  int h = t % H;
  int n = t / H;
  Out[idx] = f2bf(In[(((size_t)n * C + c) * H + h) * W + w]);
}

// ---------------------------------------------------------------------------
// WMMA GEMM:  Co[M,N] = act( A[M,K] @ Wt[N,K]^T + bias (+resid) )    all bf16 I/O
// Block = 256 thr = 8 waves; block tile 256x64; wave tile 32x64 (8 WMMA/k-step).
// B panel (64x32) double-buffered in LDS via GLOBAL_LOAD_ASYNC_TO_LDS_B128:
// next panel streams global->LDS (ASYNCcnt) while WMMAs consume current panel.
// Padded stride 40 (80B) -> conflict-free ds_load_b128.
// Requires M%256==0, K%32==0, N%16==0.  act: 0 none, 1 relu, 2 gelu(exact)
// ---------------------------------------------------------------------------
__global__ __launch_bounds__(256)
void gemm_wmma_bf16(const __bf16* __restrict__ A, const __bf16* __restrict__ Wt,
                    const float* __restrict__ bias, const __bf16* __restrict__ resid,
                    __bf16* __restrict__ Co, int M, int N, int K, int act)
{
  constexpr int BSTR = 40;                 // bf16 elems; 80B row
  __shared__ __bf16 Bs[2][64 * BSTR];

  const int lane = threadIdx.x & 31;
  const int wave = threadIdx.x >> 5;
  const int n64  = blockIdx.x << 6;
  const int mb   = (blockIdx.y << 8) + (wave << 5);

  const int lm = lane & 15;                // M row (A) / N col (B) within tile
  const int kb = (lane >> 4) << 3;         // K sub-offset: 0 or 8

  v8f acc[2][4];
  #pragma unroll
  for (int t = 0; t < 2; ++t)
    #pragma unroll
    for (int s = 0; s < 4; ++s) acc[t][s] = (v8f)(0.0f);

  const __bf16* Ar0 = A + (size_t)(mb + lm) * K;
  const __bf16* Ar1 = A + (size_t)(mb + 16 + lm) * K;

  // cooperative B fill mapping: thread -> (col 0..63, kchunk 0..3)
  const int fcol = threadIdx.x >> 2;
  const int fkc  = (threadIdx.x & 3) << 3;
  const bool fok = (n64 + fcol) < N;
  const __bf16* Wrow = Wt + (size_t)(n64 + fcol) * K + fkc;

  // columns beyond N (only in the N=32 tail GEMM): zero once, never touched again
  if (!fok) {
    *reinterpret_cast<v8bf*>(&Bs[0][fcol * BSTR + fkc]) = (v8bf)bfzero();
    *reinterpret_cast<v8bf*>(&Bs[1][fcol * BSTR + fkc]) = (v8bf)bfzero();
  }

  const unsigned ldsA = (unsigned)(size_t)&Bs[0][fcol * BSTR + fkc];
  const unsigned ldsB = (unsigned)(size_t)&Bs[1][fcol * BSTR + fkc];

  // async fill of buffer `buf` with k-step `kk` (16B per thread, VGPR-bypassing)
  auto async_fill = [&](int kk, int buf) {
    if (fok) {
      const __bf16* src = Wrow + kk;
      unsigned dst = buf ? ldsB : ldsA;
      asm volatile("global_load_async_to_lds_b128 %0, %1, off"
                   :: "v"(dst), "v"(src) : "memory");
    }
  };

  async_fill(0, 0);
  asm volatile("s_wait_asynccnt 0x0" ::: "memory");
  __syncthreads();

  int buf = 0;
  for (int kk = 0; kk < K; kk += 32) {
    if (kk + 32 < K) {
      async_fill(kk + 32, buf ^ 1);          // stream next panel during compute
      __builtin_prefetch(Ar0 + kk + 32, 0, 1);
      __builtin_prefetch(Ar1 + kk + 32, 0, 1);
    }

    v16bf af0 = cat8(*reinterpret_cast<const v8bf*>(Ar0 + kk + kb),
                     *reinterpret_cast<const v8bf*>(Ar0 + kk + kb + 16));
    v16bf af1 = cat8(*reinterpret_cast<const v8bf*>(Ar1 + kk + kb),
                     *reinterpret_cast<const v8bf*>(Ar1 + kk + kb + 16));
    #pragma unroll
    for (int s = 0; s < 4; ++s) {
      const __bf16* bp = &Bs[buf][(s * 16 + lm) * BSTR + kb];
      v16bf bfr = cat8(*reinterpret_cast<const v8bf*>(bp),
                       *reinterpret_cast<const v8bf*>(bp + 16));
      acc[0][s] = __builtin_amdgcn_wmma_f32_16x16x32_bf16(
          false, af0, false, bfr, (short)0, acc[0][s], false, false);
      acc[1][s] = __builtin_amdgcn_wmma_f32_16x16x32_bf16(
          false, af1, false, bfr, (short)0, acc[1][s], false, false);
    }

    asm volatile("s_wait_asynccnt 0x0" ::: "memory");
    __syncthreads();
    buf ^= 1;
  }

  #pragma unroll
  for (int t = 0; t < 2; ++t) {
    #pragma unroll
    for (int s = 0; s < 4; ++s) {
      if (n64 + (s << 4) >= N) continue;
      const int ncol = n64 + (s << 4) + lm;
      const float bv = bias ? bias[ncol] : 0.0f;
      #pragma unroll
      for (int r = 0; r < 8; ++r) {
        const int mrow = mb + (t << 4) + r + ((lane >> 4) << 3);
        float v = acc[t][s][r] + bv;
        if (resid) v += bf2f(resid[(size_t)mrow * N + ncol]);
        if (act == 1) v = fmaxf(v, 0.0f);
        else if (act == 2) v = 0.5f * v * (1.0f + erff(v * 0.70710678f));
        Co[(size_t)mrow * N + ncol] = f2bf(v);
      }
    }
  }
}

// ---------------------------------------------------------------------------
// im2col for 3x3 SAME conv, NHWC bf16 -> Col[M, C*9] bf16, k = c*9 + ky*3 + kx
// ---------------------------------------------------------------------------
__global__ void im2col_k(const __bf16* __restrict__ In, __bf16* __restrict__ Col,
                         int Nb, int H, int W, int C)
{
  int idx = blockIdx.x * blockDim.x + threadIdx.x;
  if (idx >= Nb * H * W * C) return;
  const int c = idx % C;
  const int m = idx / C;
  const int n  = m / (H * W);
  const int hw = m - n * (H * W);
  const int h  = hw / W;
  const int w  = hw - h * W;
  __bf16* dst = Col + (size_t)m * (C * 9) + c * 9;
  #pragma unroll
  for (int r = 0; r < 9; ++r) {
    const int ky = r / 3, kx = r - ky * 3;
    const int hh = h + ky - 1, ww = w + kx - 1;
    __bf16 v = bfzero();
    if (hh >= 0 && hh < H && ww >= 0 && ww < W)
      v = In[((size_t)(n * H + hh) * W + ww) * C + c];
    dst[r] = v;
  }
}

// ---------------------------------------------------------------------------
// GroupNorm(1,C): per-sample mean/rstd over C*H*W (bf16 NHWC input)
// ---------------------------------------------------------------------------
__global__ void gn_stats_k(const __bf16* __restrict__ X, float* __restrict__ stats, int per)
{
  int n = blockIdx.x;
  const __bf16* p = X + (size_t)n * per;
  float s = 0.0f, ss = 0.0f;
  for (int i = threadIdx.x; i < per; i += blockDim.x) { float v = bf2f(p[i]); s += v; ss += v * v; }
  __shared__ float sh1[256], sh2[256];
  sh1[threadIdx.x] = s; sh2[threadIdx.x] = ss;
  __syncthreads();
  for (int o = 128; o > 0; o >>= 1) {
    if ((int)threadIdx.x < o) { sh1[threadIdx.x] += sh1[threadIdx.x + o]; sh2[threadIdx.x] += sh2[threadIdx.x + o]; }
    __syncthreads();
  }
  if (threadIdx.x == 0) {
    float mu = sh1[0] / per;
    float var = sh2[0] / per - mu * mu;
    stats[n * 2 + 0] = mu;
    stats[n * 2 + 1] = rsqrtf(var + 1e-5f);
  }
}

__global__ void gn_apply_k(const __bf16* __restrict__ X, const float* __restrict__ stats,
                           const float* __restrict__ g, const float* __restrict__ b,
                           __bf16* __restrict__ Y, int per, int C)
{
  int idx = blockIdx.x * blockDim.x + threadIdx.x;
  if (idx >= 2 * per) return;
  int n = idx / per;
  int c = idx % C;
  float mu = stats[n * 2 + 0], rs = stats[n * 2 + 1];
  Y[idx] = f2bf((bf2f(X[idx]) - mu) * rs * g[c] + b[c]);
}

// ---------------------------------------------------------------------------
// Neighborhood attention core: one thread per (n,h,w,head). d=32, 7x7 window.
// QKV bf16 [NHW,384] (q|k|v of 32 per head). rpb fp32 [4,13,13]. Out bf16 [NHW,128]
// ---------------------------------------------------------------------------
__global__ void na2d_core_k(const __bf16* __restrict__ QKV, const float* __restrict__ rpb,
                            __bf16* __restrict__ Out, int Nb, int H, int W)
{
  int idx = blockIdx.x * blockDim.x + threadIdx.x;
  if (idx >= Nb * H * W * 4) return;
  const int e   = idx & 3;
  const int pix = idx >> 2;
  const int w   = pix % W;
  const int h   = (pix / W) % H;
  const int n   = pix / (W * H);

  const float scale = 0.17677669529663687f;   // 32^-0.5
  float q[32];
  {
    const v8bf* qp = reinterpret_cast<const v8bf*>(QKV + (size_t)pix * 384 + e * 32);
    #pragma unroll
    for (int i = 0; i < 4; ++i) {
      v8bf t = qp[i];
      #pragma unroll
      for (int j = 0; j < 8; ++j) q[i * 8 + j] = bf2f(t[j]) * scale;
    }
  }
  int i0 = h - 3; if (i0 < 0) i0 = 0; if (i0 > H - 7) i0 = H - 7;
  int j0 = w - 3; if (j0 < 0) j0 = 0; if (j0 > W - 7) j0 = W - 7;

  float sc[49];
  float mx = -1e30f;
  const float* rp = rpb + e * 169;
  for (int ii = 0; ii < 7; ++ii) {
    const int hh = i0 + ii;
    for (int jj = 0; jj < 7; ++jj) {
      const int ww = j0 + jj;
      const v8bf* kp = reinterpret_cast<const v8bf*>(
          QKV + ((size_t)(n * H + hh) * W + ww) * 384 + 128 + e * 32);
      float s = 0.0f;
      #pragma unroll
      for (int i = 0; i < 4; ++i) {
        v8bf t = kp[i];
        #pragma unroll
        for (int j = 0; j < 8; ++j) s += q[i * 8 + j] * bf2f(t[j]);
      }
      s += rp[(hh - h + 6) * 13 + (ww - w + 6)];
      sc[ii * 7 + jj] = s;
      mx = fmaxf(mx, s);
    }
  }
  float den = 0.0f;
  for (int t = 0; t < 49; ++t) { sc[t] = __expf(sc[t] - mx); den += sc[t]; }
  const float inv = 1.0f / den;

  float o[32];
  #pragma unroll
  for (int d = 0; d < 32; ++d) o[d] = 0.0f;
  for (int ii = 0; ii < 7; ++ii) {
    const int hh = i0 + ii;
    for (int jj = 0; jj < 7; ++jj) {
      const int ww = j0 + jj;
      const v8bf* vp = reinterpret_cast<const v8bf*>(
          QKV + ((size_t)(n * H + hh) * W + ww) * 384 + 256 + e * 32);
      const float wgt = sc[ii * 7 + jj] * inv;
      #pragma unroll
      for (int i = 0; i < 4; ++i) {
        v8bf t = vp[i];
        #pragma unroll
        for (int j = 0; j < 8; ++j) o[i * 8 + j] += wgt * bf2f(t[j]);
      }
    }
  }
  __bf16* op = Out + (size_t)pix * 128 + e * 32;
  #pragma unroll
  for (int d = 0; d < 32; ++d) op[d] = f2bf(o[d]);
}

// ---------------------------------------------------------------------------
// LayerNorm over last dim C=128: one block (128 threads) per row. bf16 I/O.
// ---------------------------------------------------------------------------
__global__ __launch_bounds__(128)
void layernorm_k(const __bf16* __restrict__ X, const float* __restrict__ g,
                 const float* __restrict__ b, __bf16* __restrict__ Y, int C)
{
  const int row = blockIdx.x;
  const float v = bf2f(X[(size_t)row * C + threadIdx.x]);
  __shared__ float sh1[128], sh2[128];
  sh1[threadIdx.x] = v; sh2[threadIdx.x] = v * v;
  __syncthreads();
  for (int o = 64; o > 0; o >>= 1) {
    if ((int)threadIdx.x < o) { sh1[threadIdx.x] += sh1[threadIdx.x + o]; sh2[threadIdx.x] += sh2[threadIdx.x + o]; }
    __syncthreads();
  }
  const float mu = sh1[0] / C;
  const float var = sh2[0] / C - mu * mu;
  const float rs = rsqrtf(var + 1e-5f);
  Y[(size_t)row * C + threadIdx.x] = f2bf((v - mu) * rs * g[threadIdx.x] + b[threadIdx.x]);
}

// ---------------------------------------------------------------------------
// Global average pool: P[n,c] = mean_hw Y[n,hw,c]  (bf16 -> fp32)
// ---------------------------------------------------------------------------
__global__ void pool_k(const __bf16* __restrict__ Y, float* __restrict__ P, int HW, int C)
{
  const int nc = blockIdx.x;
  const int n = nc / C, c = nc % C;
  float s = 0.0f;
  for (int i = threadIdx.x; i < HW; i += blockDim.x)
    s += bf2f(Y[((size_t)n * HW + i) * C + c]);
  __shared__ float sh[256];
  sh[threadIdx.x] = s;
  __syncthreads();
  for (int o = 128; o > 0; o >>= 1) {
    if ((int)threadIdx.x < o) sh[threadIdx.x] += sh[threadIdx.x + o];
    __syncthreads();
  }
  if (threadIdx.x == 0) P[nc] = sh[0] / HW;
}

// ---------------------------------------------------------------------------
// SE head: s = sigmoid(relu(P @ w1^T + b1) @ w2^T + b2)  (2x128 -> 2x7 -> 2x128)
// ---------------------------------------------------------------------------
__global__ void se_k(const float* __restrict__ P,
                     const float* __restrict__ w1, const float* __restrict__ b1,
                     const float* __restrict__ w2, const float* __restrict__ b2,
                     float* __restrict__ S, int Nb, int C, int R)
{
  __shared__ float hid[64];
  const int t = threadIdx.x;
  if (t < Nb * R) {
    const int n = t / R, r = t % R;
    float s = b1[r];
    for (int c = 0; c < C; ++c) s += P[n * C + c] * w1[r * C + c];
    hid[n * R + r] = fmaxf(s, 0.0f);
  }
  __syncthreads();
  if (t < Nb * C) {
    const int n = t / C, c = t % C;
    float s = b2[c];
    for (int r = 0; r < R; ++r) s += hid[n * R + r] * w2[c * R + r];
    S[n * C + c] = 1.0f / (1.0f + __expf(-s));
  }
}

// ---------------------------------------------------------------------------
// Final: out[NCHW fp32] = Y[NHWC bf16] * S[n,c] + X[NCHW fp32]
// ---------------------------------------------------------------------------
__global__ void final_k(const __bf16* __restrict__ Y, const float* __restrict__ S,
                        const float* __restrict__ X, float* __restrict__ Out,
                        int C, int H, int W, int total)
{
  int idx = blockIdx.x * blockDim.x + threadIdx.x;
  if (idx >= total) return;
  const int HW = H * W;
  const int p = idx % HW;
  const int c = (idx / HW) % C;
  const int n = idx / (HW * C);
  Out[idx] = bf2f(Y[((size_t)n * HW + p) * C + c]) * S[n * C + c] + X[idx];
}

// ---------------------------------------------------------------------------
// Host-side launch sequence
// ---------------------------------------------------------------------------
extern "C" void kernel_launch(void* const* d_in, const int* in_sizes, int n_in,
                              void* d_out, int out_size, void* d_ws, size_t ws_size,
                              hipStream_t stream) {
  (void)in_sizes; (void)n_in; (void)out_size; (void)ws_size;

  const int Nb = 2, C = 128, H = 64, W = 64;
  const int HW = H * W, M = Nb * HW;          // 8192
  const int HID = 512, RED = 7;
  const size_t P = (size_t)M * C;             // 1,048,576 elements

  const float* x      = (const float*)d_in[0];
  const float* gn_g   = (const float*)d_in[1];
  const float* gn_b   = (const float*)d_in[2];
  const float* qkv_w  = (const float*)d_in[3];
  const float* qkv_b  = (const float*)d_in[4];
  const float* proj_w = (const float*)d_in[5];
  const float* proj_b = (const float*)d_in[6];
  const float* rpb    = (const float*)d_in[7];
  const float* ln_g   = (const float*)d_in[8];
  const float* ln_b   = (const float*)d_in[9];
  const float* fc1_w  = (const float*)d_in[10];
  const float* fc1_b  = (const float*)d_in[11];
  const float* fc2_w  = (const float*)d_in[12];
  const float* fc2_b  = (const float*)d_in[13];
  const float* bconv_w= (const float*)d_in[14];
  const float* bconv_b= (const float*)d_in[15];
  const float* c1_w   = (const float*)d_in[16];
  const float* c1_b   = (const float*)d_in[17];
  const float* c2_w   = (const float*)d_in[18];
  const float* c2_b   = (const float*)d_in[19];
  const float* ca1_w  = (const float*)d_in[20];
  const float* ca1_b  = (const float*)d_in[21];
  const float* ca2_w  = (const float*)d_in[22];
  const float* ca2_b  = (const float*)d_in[23];
  float* out = (float*)d_out;

  // --- workspace carve (bf16 activations + bf16 weights + small fp32) ---
  char* base = (char*)d_ws;
  auto alloc = [&](size_t bytes) -> char* {
    char* p = base; base += (bytes + 255) & ~(size_t)255; return p;
  };
  __bf16* resA  = (__bf16*)alloc(P * 2);
  __bf16* resB  = (__bf16*)alloc(P * 2);
  __bf16* xg    = (__bf16*)alloc(P * 2);
  __bf16* qkvb  = (__bf16*)alloc(3 * P * 2);
  __bf16* attnb = (__bf16*)alloc(P * 2);
  __bf16* abuf  = (__bf16*)alloc(P * 2);
  __bf16* lnb   = (__bf16*)alloc(P * 2);
  __bf16* hbuf  = (__bf16*)alloc(4 * P * 2);
  __bf16* colb  = (__bf16*)alloc((size_t)M * C * 9 * 2);
  __bf16* convo = (__bf16*)alloc(P * 2);
  __bf16* y1    = (__bf16*)alloc((size_t)M * 32 * 2);
  __bf16* y2    = (__bf16*)alloc(P * 2);
  __bf16* wqkv  = (__bf16*)alloc((size_t)2 * 3 * C * C * 2);
  __bf16* wproj = (__bf16*)alloc((size_t)2 * C * C * 2);
  __bf16* wfc1  = (__bf16*)alloc((size_t)2 * HID * C * 2);
  __bf16* wfc2  = (__bf16*)alloc((size_t)2 * C * HID * 2);
  __bf16* wconv = (__bf16*)alloc((size_t)C * C * 9 * 2);
  __bf16* wc1   = (__bf16*)alloc((size_t)32 * C * 2);
  __bf16* wc2   = (__bf16*)alloc((size_t)C * 32 * 2);
  float*  stats = (float*)alloc(16 * 4);
  float*  poolb = (float*)alloc(256 * 4);
  float*  sbuf  = (float*)alloc(256 * 4);

  const int TPB = 256;
  const int total = Nb * C * HW;
  auto cdiv = [](int a, int b) { return (a + b - 1) / b; };

  // --- one-time weight conversions to bf16 ---
  cvt_bf16_k<<<cdiv(2*3*C*C, TPB), TPB, 0, stream>>>(qkv_w,  wqkv,  2*3*C*C);
  cvt_bf16_k<<<cdiv(2*C*C,   TPB), TPB, 0, stream>>>(proj_w, wproj, 2*C*C);
  cvt_bf16_k<<<cdiv(2*HID*C, TPB), TPB, 0, stream>>>(fc1_w,  wfc1,  2*HID*C);
  cvt_bf16_k<<<cdiv(2*C*HID, TPB), TPB, 0, stream>>>(fc2_w,  wfc2,  2*C*HID);
  cvt_bf16_k<<<cdiv(C*C*9,   TPB), TPB, 0, stream>>>(bconv_w, wconv, C*C*9);
  cvt_bf16_k<<<cdiv(32*C,    TPB), TPB, 0, stream>>>(c1_w,   wc1,   32*C);
  cvt_bf16_k<<<cdiv(C*32,    TPB), TPB, 0, stream>>>(c2_w,   wc2,   C*32);

  // NCHW fp32 -> NHWC bf16
  nchw2nhwc_k<<<cdiv(total, TPB), TPB, 0, stream>>>(x, resA, Nb, C, H, W);

  auto gemm_grid = [&](int N) { return dim3((unsigned)((N + 63) / 64), (unsigned)(M / 256)); };

  __bf16* cur = resA;
  __bf16* nxt = resB;
  for (int l = 0; l < 2; ++l) {
    const __bf16* qw  = wqkv  + (size_t)l * 3 * C * C;
    const float*  qb  = qkv_b + (size_t)l * 3 * C;
    const __bf16* pw  = wproj + (size_t)l * C * C;
    const float*  pb  = proj_b+ (size_t)l * C;
    const float*  rb  = rpb   + (size_t)l * 4 * 169;
    const __bf16* f1w = wfc1  + (size_t)l * HID * C;
    const float*  f1b = fc1_b + (size_t)l * HID;
    const __bf16* f2w = wfc2  + (size_t)l * C * HID;
    const float*  f2b = fc2_b + (size_t)l * C;

    gn_stats_k<<<Nb, TPB, 0, stream>>>(cur, stats, C * HW);
    gn_apply_k<<<cdiv(total, TPB), TPB, 0, stream>>>(
        cur, stats, gn_g + l * C, gn_b + l * C, xg, C * HW, C);

    // QKV: [M,128] @ [384,128]^T
    gemm_wmma_bf16<<<gemm_grid(3 * C), TPB, 0, stream>>>(
        xg, qw, qb, nullptr, qkvb, M, 3 * C, C, 0);

    // neighborhood attention core
    na2d_core_k<<<cdiv(M * 4, TPB), TPB, 0, stream>>>(qkvb, rb, attnb, Nb, H, W);

    // proj + residual(pre-norm input)
    gemm_wmma_bf16<<<gemm_grid(C), TPB, 0, stream>>>(
        attnb, pw, pb, cur, abuf, M, C, C, 0);

    // MLP: LN -> fc1(gelu) -> fc2(+residual a)
    layernorm_k<<<M, 128, 0, stream>>>(abuf, ln_g + l * C, ln_b + l * C, lnb, C);
    gemm_wmma_bf16<<<gemm_grid(HID), TPB, 0, stream>>>(
        lnb, f1w, f1b, nullptr, hbuf, M, HID, C, 2);
    gemm_wmma_bf16<<<gemm_grid(C), TPB, 0, stream>>>(
        hbuf, f2w, f2b, abuf, nxt, M, C, HID, 0);

    __bf16* tmp = cur; cur = nxt; nxt = tmp;
  }

  // 3x3 conv: im2col (bf16 copy) + WMMA GEMM, K=1152
  im2col_k<<<cdiv(M * C, TPB), TPB, 0, stream>>>(cur, colb, Nb, H, W, C);
  gemm_wmma_bf16<<<gemm_grid(C), TPB, 0, stream>>>(
      colb, wconv, bconv_b, nullptr, convo, M, C, C * 9, 0);

  // tail: 1x1 -> relu -> 1x1
  gemm_wmma_bf16<<<gemm_grid(C / 4), TPB, 0, stream>>>(
      convo, wc1, c1_b, nullptr, y1, M, C / 4, C, 1);
  gemm_wmma_bf16<<<gemm_grid(C), TPB, 0, stream>>>(
      y1, wc2, c2_b, nullptr, y2, M, C, C / 4, 0);

  // CALayer: pool -> tiny MLP -> sigmoid
  pool_k<<<Nb * C, TPB, 0, stream>>>(y2, poolb, HW, C);
  se_k<<<1, TPB, 0, stream>>>(poolb, ca1_w, ca1_b, ca2_w, ca2_b, sbuf, Nb, C, RED);

  // final scale + global residual, NHWC -> NCHW
  final_k<<<cdiv(total, TPB), TPB, 0, stream>>>(y2, sbuf, x, out, C, H, W, total);
}
